// InternVisionEncoderLayer_31688268710156
// MI455X (gfx1250) — matrix operand order repaired and checked by reading, not compiled, not run on hardware
//
#include <hip/hip_runtime.h>
#include <hip/hip_bf16.h>
#include <math.h>

#define S_TOK  2048
#define NSEQQ  4
#define SEQL   (S_TOK / NSEQQ)   // 512
#define E_DIM  1024
#define H_HEADS 16
#define D_HEAD 64
#define I_DIM  4096
#define EPSF   1e-6f

typedef __attribute__((ext_vector_type(16))) _Float16 v16h;
typedef __attribute__((ext_vector_type(8)))  _Float16 v8h;
typedef __attribute__((ext_vector_type(8)))  float    v8f;

static __device__ __forceinline__ v16h mk16(v8h a, v8h b) {
  union { v16h v; v8h h[2]; } u;
  u.h[0] = a; u.h[1] = b;
  return u.v;
}

// low 32 bits of a generic pointer to LDS == workgroup-relative LDS byte addr
static __device__ __forceinline__ unsigned lds_lo32(const void* p) {
  return (unsigned)(uintptr_t)p;
}

// ---------------------------------------------------------------------------
// f32 -> f16 conversion (weights, grid-stride)
// ---------------------------------------------------------------------------
__global__ void cvt_f32_to_f16(const float* __restrict__ in,
                               _Float16* __restrict__ out, int n) {
  for (int i = blockIdx.x * blockDim.x + threadIdx.x; i < n;
       i += gridDim.x * blockDim.x)
    out[i] = (_Float16)in[i];
}

// ---------------------------------------------------------------------------
// LayerNorm (f32 in) -> f16 out, one block per row of E=1024
// ---------------------------------------------------------------------------
__global__ __launch_bounds__(256) void layernorm_to_f16(
    const float* __restrict__ x, const float* __restrict__ w,
    const float* __restrict__ b, _Float16* __restrict__ y) {
  __shared__ float ssum[256], ssq[256];
  const int row = blockIdx.x;
  const int tid = threadIdx.x;
  const float* xr = x + (size_t)row * E_DIM;
  float s = 0.f, q = 0.f;
  for (int e = tid; e < E_DIM; e += 256) {
    float v = xr[e];
    s += v; q += v * v;
  }
  ssum[tid] = s; ssq[tid] = q;
  __syncthreads();
  for (int st = 128; st > 0; st >>= 1) {
    if (tid < st) { ssum[tid] += ssum[tid + st]; ssq[tid] += ssq[tid + st]; }
    __syncthreads();
  }
  const float mu  = ssum[0] * (1.0f / E_DIM);
  const float var = ssq[0] * (1.0f / E_DIM) - mu * mu;
  const float rs  = rsqrtf(var + EPSF);
  for (int e = tid; e < E_DIM; e += 256) {
    float v = (xr[e] - mu) * rs * w[e] + b[e];
    y[(size_t)row * E_DIM + e] = (_Float16)v;
  }
}

// ---------------------------------------------------------------------------
// Tiled WMMA GEMM:  C[M,N] = A[M,K] * B[N,K]^T + bias, NT layout.
// Block tile 128x128, BK=32, double-buffered LDS filled by
// global_load_async_to_lds_b128 (ASYNCcnt). 8 waves, each 32x64
// (2 A-frags x 4 B-frags = 8 wmma per K-step).
// EPI: 0 = store f32;  1 = f16(gelu);  2 = f32 resid + v*ls (layer-scale)
// M % 128 == 0, N % 128 == 0, K % 32 == 0 (true for all four GEMMs here).
// ---------------------------------------------------------------------------
template <int EPI>
__global__ __launch_bounds__(256) void gemm_nt_wmma(
    const _Float16* __restrict__ A, const _Float16* __restrict__ B,
    const float* __restrict__ bias, const float* __restrict__ ls,
    const float* __restrict__ resid, float* __restrict__ Cf,
    _Float16* __restrict__ Ch, int M, int N, int K) {
  constexpr int BM = 128, BN = 128, BK = 32, BKP = 40;  // +8 halves pad
  __shared__ _Float16 As[2][BM][BKP];
  __shared__ _Float16 Bs[2][BN][BKP];

  const int tid  = threadIdx.x;
  const int lane = tid & 31;
  const int wave = tid >> 5;
  const int wm   = wave >> 1;  // 0..3  (rows, 32 each)
  const int wn   = wave & 1;   // 0..1  (cols, 64 each)
  const int lr   = lane & 15;
  const int hi   = lane >> 4;

  const int n0 = blockIdx.x * BN;
  const int m0 = blockIdx.y * BM;

  // per-thread copy slice: 32B of A + 32B of B per K-step
  const int arow = tid >> 1;        // 0..127
  const int aseg = (tid & 1) * 16;  // halves: 0/16
  const int brow = tid >> 1;        // 0..127
  const int bseg = (tid & 1) * 16;

  const unsigned long long abase = (unsigned long long)A;
  const unsigned long long bbase = (unsigned long long)B;
  const unsigned a_goff = ((unsigned)(m0 + arow) * (unsigned)K + aseg) * 2u;
  const unsigned b_goff = ((unsigned)(n0 + brow) * (unsigned)K + bseg) * 2u;

  // issue 4x async b128 global->LDS copies for one K-tile (GVS mode)
  auto issue_tile = [&](int buf, unsigned kbytes) {
    unsigned a_l = lds_lo32(&As[buf][arow][aseg]);
    unsigned b_l = lds_lo32(&Bs[buf][brow][bseg]);
    unsigned a_g = a_goff + kbytes;
    unsigned b_g = b_goff + kbytes;
    asm volatile(
        "global_load_async_to_lds_b128 %0, %4, %8\n\t"
        "global_load_async_to_lds_b128 %1, %5, %8\n\t"
        "global_load_async_to_lds_b128 %2, %6, %9\n\t"
        "global_load_async_to_lds_b128 %3, %7, %9"
        :
        : "v"(a_l), "v"(a_l + 16u), "v"(b_l), "v"(b_l + 16u),
          "v"(a_g), "v"(a_g + 16u), "v"(b_g), "v"(b_g + 16u),
          "s"(abase), "s"(bbase)
        : "memory");
  };

  v8f acc[2][4] = {};

  issue_tile(0, 0u);
  asm volatile("s_wait_asynccnt 0x0" ::: "memory");
  __syncthreads();

  const int nsteps = K / BK;
  for (int t = 0; t < nsteps; t++) {
    const int buf = t & 1;
    if (t + 1 < nsteps) issue_tile(buf ^ 1, (unsigned)(t + 1) * (BK * 2u));

    // fragments per ISA wave32 layouts
    v16h af[2], bf[4];
#pragma unroll
    for (int i = 0; i < 2; i++) {
      const int r = wm * 32 + i * 16 + lr;
      v8h lo = *(const v8h*)&As[buf][r][hi * 8];       // K = hi*8 .. +7
      v8h h2 = *(const v8h*)&As[buf][r][16 + hi * 8];  // K = 16+hi*8 .. +7
      af[i] = mk16(lo, h2);
    }
#pragma unroll
    for (int j = 0; j < 4; j++) {
      const int c = wn * 64 + j * 16 + lr;
      v8h lo = *(const v8h*)&Bs[buf][c][hi * 8];       // K = hi*8 .. +7
      v8h h2 = *(const v8h*)&Bs[buf][c][16 + hi * 8];  // K = 16+hi*8 .. +7
      bf[j] = mk16(lo, h2);
    }
#pragma unroll
    for (int i = 0; i < 2; i++)
#pragma unroll
      for (int j = 0; j < 4; j++)
        acc[i][j] = __builtin_amdgcn_wmma_f32_16x16x32_f16(
            false, af[i], false, bf[j], (short)0, acc[i][j], false, false);

    if (t + 1 < nsteps) asm volatile("s_wait_asynccnt 0x0" ::: "memory");
    __syncthreads();
  }

  // epilogue (C layout: lane lr = col, rows hi*8 + r)
#pragma unroll
  for (int i = 0; i < 2; i++) {
#pragma unroll
    for (int j = 0; j < 4; j++) {
      const int col  = n0 + wn * 64 + j * 16 + lr;
      const float bc = bias[col];
#pragma unroll
      for (int r = 0; r < 8; r++) {
        const int row = m0 + wm * 32 + i * 16 + hi * 8 + r;
        float v = acc[i][j][r] + bc;
        if constexpr (EPI == 0) {
          Cf[(size_t)row * N + col] = v;
        } else if constexpr (EPI == 1) {
          float g = 0.5f * v * (1.0f + erff(v * 0.70710678118f));
          Ch[(size_t)row * N + col] = (_Float16)g;
        } else {
          Cf[(size_t)row * N + col] =
              resid[(size_t)row * N + col] + v * ls[col];
        }
      }
    }
  }
}

// ---------------------------------------------------------------------------
// QKV post: joint-E RMS-norm of q/k rows + RoPE + f16 pack, one block per s.
// qkv is [S, 3, E] f32; outputs q/k/v as [S, E] f16.
// ---------------------------------------------------------------------------
__global__ __launch_bounds__(256) void qkv_post(
    const float* __restrict__ qkv, const float* __restrict__ rope,
    const float* __restrict__ qw, const float* __restrict__ kw,
    _Float16* __restrict__ qh, _Float16* __restrict__ kh,
    _Float16* __restrict__ vh) {
  __shared__ float sq[256], sk[256];
  const int s   = blockIdx.x;
  const int tid = threadIdx.x;
  const float* base = qkv + (size_t)s * 3 * E_DIM;
  float aq = 0.f, ak = 0.f;
  for (int e = tid; e < E_DIM; e += 256) {
    float q = base[e], k = base[E_DIM + e];
    aq += q * q; ak += k * k;
  }
  sq[tid] = aq; sk[tid] = ak;
  __syncthreads();
  for (int st = 128; st > 0; st >>= 1) {
    if (tid < st) { sq[tid] += sq[tid + st]; sk[tid] += sk[tid + st]; }
    __syncthreads();
  }
  const float rq = rsqrtf(sq[0] * (1.0f / E_DIM) + EPSF);
  const float rk = rsqrtf(sk[0] * (1.0f / E_DIM) + EPSF);

  for (int e = tid; e < E_DIM; e += 256) {
    const int d    = e & (D_HEAD - 1);
    const int dh   = d & 31;                       // d mod (D/2)
    const int pair = (d < 32) ? (e + 32) : (e - 32);
    const float f  = rope[(size_t)s * 32 + dh];
    const float c  = cosf(f), sn = sinf(f);
    // q: rms-norm then rope
    float yq   = base[e]    * rq * qw[e];
    float yqp  = base[pair] * rq * qw[pair];
    float rotq = (d < 32) ? -yqp : yqp;
    qh[(size_t)s * E_DIM + e] = (_Float16)(yq * c + rotq * sn);
    // k
    float yk   = base[E_DIM + e]    * rk * kw[e];
    float ykp  = base[E_DIM + pair] * rk * kw[pair];
    float rotk = (d < 32) ? -ykp : ykp;
    kh[(size_t)s * E_DIM + e] = (_Float16)(yk * c + rotk * sn);
    // v: pass-through to f16
    vh[(size_t)s * E_DIM + e] = (_Float16)base[2 * E_DIM + e];
  }
}

// ---------------------------------------------------------------------------
// Flash attention, block-diagonal (4 segments of 512), D=64, H=16.
// Grid: (NSEQ * 4 q-blocks, H). Block: 256 thr = 8 waves, each wave 16 q-rows.
// Scores and P*V both on v_wmma_f32_16x16x32_f16; online softmax via shfl_xor.
// ---------------------------------------------------------------------------
__global__ __launch_bounds__(256) void attn_flash(
    const _Float16* __restrict__ Q, const _Float16* __restrict__ Kh,
    const _Float16* __restrict__ V, _Float16* __restrict__ O) {
  __shared__ _Float16 Ks[64][72];       // [kv][d]
  __shared__ _Float16 Vt[64][72];       // [d][kv]  (transposed for B frags)
  __shared__ _Float16 Ps[8][16][72];    // per-wave P staging [qrow][kv]

  const int tid  = threadIdx.x;
  const int lane = tid & 31;
  const int wave = tid >> 5;
  const int lr   = lane & 15;
  const int hi   = lane >> 4;

  const int h    = blockIdx.y;
  const int seq  = blockIdx.x >> 2;
  const int qb   = blockIdx.x & 3;
  const int q0   = seq * SEQL + qb * 128 + wave * 16;
  const int hoff = h * D_HEAD;

  // preload Q fragments for this wave's 16 rows (2 k-steps over D=64)
  const _Float16* qbase = Q + (size_t)(q0 + lr) * E_DIM + hoff;
  v16h qa[2];
#pragma unroll
  for (int ks = 0; ks < 2; ks++) {
    v8h lo = *(const v8h*)(qbase + ks * 32 + hi * 8);
    v8h h2 = *(const v8h*)(qbase + ks * 32 + 16 + hi * 8);
    qa[ks] = mk16(lo, h2);
  }

  float m_r[8], l_r[8];
  v8f o_acc[4] = {};
#pragma unroll
  for (int r = 0; r < 8; r++) { m_r[r] = -1e30f; l_r[r] = 0.0f; }

  const int   krow  = tid >> 2;
  const int   kseg  = (tid & 3) * 16;
  const float scale = 0.125f;  // 1/sqrt(64)

  for (int t = 0; t < SEQL / 64; t++) {
    const int kv0 = seq * SEQL + t * 64;
    v8h ka = *(const v8h*)(Kh + (size_t)(kv0 + krow) * E_DIM + hoff + kseg);
    v8h kb = *(const v8h*)(Kh + (size_t)(kv0 + krow) * E_DIM + hoff + kseg + 8);
    v8h va = *(const v8h*)(V  + (size_t)(kv0 + krow) * E_DIM + hoff + kseg);
    v8h vb = *(const v8h*)(V  + (size_t)(kv0 + krow) * E_DIM + hoff + kseg + 8);
    __syncthreads();
    *(v8h*)&Ks[krow][kseg]     = ka;
    *(v8h*)&Ks[krow][kseg + 8] = kb;
#pragma unroll
    for (int u = 0; u < 8; u++) {   // transpose V into Vt
      Vt[kseg + u][krow]     = va[u];
      Vt[kseg + 8 + u][krow] = vb[u];
    }
    __syncthreads();

    // scores: 16 q-rows x 64 kv  (4 N-tiles x 2 k-steps of WMMA)
    float sc[4][8];
#pragma unroll
    for (int nt = 0; nt < 4; nt++) {
      v8f a = {};
#pragma unroll
      for (int ks = 0; ks < 2; ks++) {
        v8h lo = *(const v8h*)&Ks[nt * 16 + lr][ks * 32 + hi * 8];
        v8h h2 = *(const v8h*)&Ks[nt * 16 + lr][ks * 32 + 16 + hi * 8];
        v16h b = mk16(lo, h2);
        a = __builtin_amdgcn_wmma_f32_16x16x32_f16(false, qa[ks], false, b,
                                                   (short)0, a, false, false);
      }
#pragma unroll
      for (int r = 0; r < 8; r++) sc[nt][r] = a[r] * scale;
    }

    // online softmax (row reductions across 16 lanes via xor shuffles)
#pragma unroll
    for (int r = 0; r < 8; r++) {
      float tm = fmaxf(fmaxf(sc[0][r], sc[1][r]), fmaxf(sc[2][r], sc[3][r]));
#pragma unroll
      for (int ml = 8; ml >= 1; ml >>= 1) tm = fmaxf(tm, __shfl_xor(tm, ml, 32));
      const float mn   = fmaxf(m_r[r], tm);
      const float corr = __expf(m_r[r] - mn);
      m_r[r] = mn;
      float ps = 0.0f;
#pragma unroll
      for (int nt = 0; nt < 4; nt++) {
        float p = __expf(sc[nt][r] - mn);
        Ps[wave][hi * 8 + r][nt * 16 + lr] = (_Float16)p;
        ps += p;
      }
#pragma unroll
      for (int ml = 8; ml >= 1; ml >>= 1) ps += __shfl_xor(ps, ml, 32);
      l_r[r] = l_r[r] * corr + ps;
#pragma unroll
      for (int j = 0; j < 4; j++) o_acc[j][r] *= corr;
    }

    // O += P x V  (A = Ps 16x64, B = Vt rows; 2 k-steps x 4 d-tiles)
#pragma unroll
    for (int ks = 0; ks < 2; ks++) {
      v8h lo = *(const v8h*)&Ps[wave][lr][ks * 32 + hi * 8];
      v8h h2 = *(const v8h*)&Ps[wave][lr][ks * 32 + 16 + hi * 8];
      v16h a = mk16(lo, h2);
#pragma unroll
      for (int j = 0; j < 4; j++) {
        v8h bl = *(const v8h*)&Vt[j * 16 + lr][ks * 32 + hi * 8];
        v8h bh = *(const v8h*)&Vt[j * 16 + lr][ks * 32 + 16 + hi * 8];
        v16h b = mk16(bl, bh);
        o_acc[j] = __builtin_amdgcn_wmma_f32_16x16x32_f16(
            false, a, false, b, (short)0, o_acc[j], false, false);
      }
    }
  }

#pragma unroll
  for (int j = 0; j < 4; j++)
#pragma unroll
    for (int r = 0; r < 8; r++) {
      float val = o_acc[j][r] / l_r[r];
      O[(size_t)(q0 + hi * 8 + r) * E_DIM + hoff + j * 16 + lr] = (_Float16)val;
    }
}

// ---------------------------------------------------------------------------
// Host-side orchestration
// ---------------------------------------------------------------------------
extern "C" void kernel_launch(void* const* d_in, const int* in_sizes, int n_in,
                              void* d_out, int out_size, void* d_ws,
                              size_t ws_size, hipStream_t stream) {
  (void)in_sizes; (void)n_in; (void)out_size; (void)ws_size;
  const float* x      = (const float*)d_in[0];
  /* d_in[1] cu_seqlens: uniform 512-token segments by construction */
  const float* rope   = (const float*)d_in[2];
  const float* qkv_w  = (const float*)d_in[3];
  const float* qkv_b  = (const float*)d_in[4];
  const float* q_nw   = (const float*)d_in[5];
  const float* k_nw   = (const float*)d_in[6];
  const float* proj_w = (const float*)d_in[7];
  const float* proj_b = (const float*)d_in[8];
  const float* n1w    = (const float*)d_in[9];
  const float* n1b    = (const float*)d_in[10];
  const float* n2w    = (const float*)d_in[11];
  const float* n2b    = (const float*)d_in[12];
  const float* fc1_w  = (const float*)d_in[13];
  const float* fc1_b  = (const float*)d_in[14];
  const float* fc2_w  = (const float*)d_in[15];
  const float* fc2_b  = (const float*)d_in[16];
  const float* ls1    = (const float*)d_in[17];
  const float* ls2    = (const float*)d_in[18];
  float* out = (float*)d_out;

  // workspace layout (256B aligned bump allocator)
  char* p = (char*)d_ws;
  auto alloc = [&](size_t bytes) -> char* {
    char* r = p;
    p += (bytes + 255) & ~(size_t)255;
    return r;
  };
  _Float16* qkvw_h = (_Float16*)alloc((size_t)3 * E_DIM * E_DIM * 2);
  _Float16* projw_h = (_Float16*)alloc((size_t)E_DIM * E_DIM * 2);
  _Float16* fc1w_h  = (_Float16*)alloc((size_t)I_DIM * E_DIM * 2);
  _Float16* fc2w_h  = (_Float16*)alloc((size_t)E_DIM * I_DIM * 2);
  _Float16* h1      = (_Float16*)alloc((size_t)S_TOK * E_DIM * 2);
  float*    qkv_f   = (float*)   alloc((size_t)S_TOK * 3 * E_DIM * 4);
  _Float16* qh      = (_Float16*)alloc((size_t)S_TOK * E_DIM * 2);
  _Float16* kh      = (_Float16*)alloc((size_t)S_TOK * E_DIM * 2);
  _Float16* vh      = (_Float16*)alloc((size_t)S_TOK * E_DIM * 2);
  _Float16* attn_h  = (_Float16*)alloc((size_t)S_TOK * E_DIM * 2);
  float*    x2      = (float*)   alloc((size_t)S_TOK * E_DIM * 4);
  _Float16* h2      = (_Float16*)alloc((size_t)S_TOK * E_DIM * 2);
  _Float16* mh      = (_Float16*)alloc((size_t)S_TOK * I_DIM * 2);

  auto cvt = [&](const float* src, _Float16* dst, int n) {
    int blocks = (n + 255) / 256;
    if (blocks > 2048) blocks = 2048;
    cvt_f32_to_f16<<<blocks, 256, 0, stream>>>(src, dst, n);
  };
  cvt(qkv_w, qkvw_h, 3 * E_DIM * E_DIM);
  cvt(proj_w, projw_h, E_DIM * E_DIM);
  cvt(fc1_w, fc1w_h, I_DIM * E_DIM);
  cvt(fc2_w, fc2w_h, E_DIM * I_DIM);

  // 1) h1 = LN1(x) in f16
  layernorm_to_f16<<<S_TOK, 256, 0, stream>>>(x, n1w, n1b, h1);

  // 2) qkv = h1 @ qkv_w^T + qkv_b   (f32 out for norm accuracy)
  gemm_nt_wmma<0><<<dim3(3 * E_DIM / 128, S_TOK / 128), 256, 0, stream>>>(
      h1, qkvw_h, qkv_b, nullptr, nullptr, qkv_f, nullptr,
      S_TOK, 3 * E_DIM, E_DIM);

  // 3) q/k RMS-norm + RoPE, v pack -> f16
  qkv_post<<<S_TOK, 256, 0, stream>>>(qkv_f, rope, q_nw, k_nw, qh, kh, vh);

  // 4) block-diagonal flash attention
  attn_flash<<<dim3(NSEQQ * 4, H_HEADS), 256, 0, stream>>>(qh, kh, vh, attn_h);

  // 5) x2 = x + (attn @ proj_w^T + proj_b) * ls1
  gemm_nt_wmma<2><<<dim3(E_DIM / 128, S_TOK / 128), 256, 0, stream>>>(
      attn_h, projw_h, proj_b, ls1, x, x2, nullptr, S_TOK, E_DIM, E_DIM);

  // 6) h2 = LN2(x2)
  layernorm_to_f16<<<S_TOK, 256, 0, stream>>>(x2, n2w, n2b, h2);

  // 7) m = gelu(h2 @ fc1_w^T + fc1_b) -> f16
  gemm_nt_wmma<1><<<dim3(I_DIM / 128, S_TOK / 128), 256, 0, stream>>>(
      h2, fc1w_h, fc1_b, nullptr, nullptr, nullptr, mh, S_TOK, I_DIM, E_DIM);

  // 8) out = x2 + (m @ fc2_w^T + fc2_b) * ls2
  gemm_nt_wmma<2><<<dim3(E_DIM / 128, S_TOK / 128), 256, 0, stream>>>(
      mh, fc2w_h, fc2_b, ls2, x2, out, nullptr, S_TOK, E_DIM, I_DIM);
}
